// KeyValRetRNNDecoder_19808389169193
// MI455X (gfx1250) — matrix operand order repaired and checked by reading, not compiled, not run on hardware
//
#include <hip/hip_runtime.h>
#include <hip/hip_bf16.h>

// ---------------------------------------------------------------------------
// Problem constants (match reference)
// ---------------------------------------------------------------------------
constexpr int BB  = 64;     // batch
constexpr int TT  = 64;     // time steps
constexpr int SS  = 128;    // source length
constexpr int EE  = 512;    // embed
constexpr int HH  = 1024;   // hidden
constexpr int VV  = 32000;  // vocab
constexpr int KDc = 64;     // KD
constexpr int DKB = 512;
constexpr int KBT = KDc * KDc;   // 4096
constexpr int NATT = 4;

// ---------------------------------------------------------------------------
// WMMA types (CDNA5: wave32, V_WMMA_F32_16X16X32_BF16)
// ---------------------------------------------------------------------------
typedef __attribute__((ext_vector_type(16))) __bf16        v16bf;
typedef __attribute__((ext_vector_type(8)))  float         v8f;
typedef __attribute__((ext_vector_type(8)))  unsigned int  v8ui;

__device__ __forceinline__ __bf16 f2bf(float x) {
  // round-to-nearest-even f32 -> bf16
  unsigned int u = __builtin_bit_cast(unsigned int, x);
  unsigned int r = (u + 0x7FFFu + ((u >> 16) & 1u)) >> 16;
  unsigned short s = (unsigned short)r;
  return __builtin_bit_cast(__bf16, s);
}

// 32 contiguous bytes (16 bf16) -> v16bf
__device__ __forceinline__ v16bf ld_b_frag(const __bf16* p) {
  uint4 lo = *reinterpret_cast<const uint4*>(p);
  uint4 hi = *reinterpret_cast<const uint4*>(p + 8);
  v8ui t;
  t[0] = lo.x; t[1] = lo.y; t[2] = lo.z; t[3] = lo.w;
  t[4] = hi.x; t[5] = hi.y; t[6] = hi.z; t[7] = hi.w;
  return __builtin_bit_cast(v16bf, t);
}

// A fragment: elems 0..7 = K base+hi*8.., elems 8..15 = K base+16+hi*8..
__device__ __forceinline__ v16bf ld_a_frag(const __bf16* p, int hi) {
  uint4 lo = *reinterpret_cast<const uint4*>(p + hi * 8);
  uint4 hi2 = *reinterpret_cast<const uint4*>(p + 16 + hi * 8);
  v8ui t;
  t[0] = lo.x;  t[1] = lo.y;  t[2] = lo.z;  t[3] = lo.w;
  t[4] = hi2.x; t[5] = hi2.y; t[6] = hi2.z; t[7] = hi2.w;
  return __builtin_bit_cast(v16bf, t);
}

// ---------------------------------------------------------------------------
// bf16 WMMA GEMM:  C[M,N] = act( A[M,K] * Bt[N,K]^T + bias )
//   A  : (M,K) bf16 row-major, lda
//   Bt : (N,K) bf16 row-major, ldb   (i.e. B pre-transposed -> contiguous K)
//   Cf : optional f32 output, Cb : optional bf16 output
// Block = 128 threads (4 waves, 2x2): block tile 64(M) x 128(N);
// wave tile 32x64 = 2 A-frags x 4 B-frags = 8 WMMAs / K-step.
// ---------------------------------------------------------------------------
__global__ __launch_bounds__(128) void wmma_gemm_bf16_kernel(
    const __bf16* __restrict__ A, const __bf16* __restrict__ Bt,
    const float* __restrict__ bias,
    float* __restrict__ Cf, __bf16* __restrict__ Cb,
    int M, int N, int K, int lda, int ldb, int ldcf, int ldcb, int act,
    long long sA, long long sB, long long sC)
{
  const int z = blockIdx.z;
  A += z * sA;  Bt += z * sB;
  if (Cf) Cf += z * sC;
  if (Cb) Cb += z * sC;

  const int wave = threadIdx.x >> 5;
  const int lane = threadIdx.x & 31;
  const int r  = lane & 15;
  const int hi = lane >> 4;
  const int row0 = blockIdx.y * 64 + (wave >> 1) * 32;
  const int col0 = blockIdx.x * 128 + (wave & 1) * 64;
  if (row0 >= M) return;

  v8f acc[2][4] = {};
  const __bf16* a0p = A + (long long)(row0 + r) * lda;
  const __bf16* a1p = A + (long long)(row0 + 16 + r) * lda;

  for (int k0 = 0; k0 < K; k0 += 32) {
    const v16bf af0 = ld_a_frag(a0p + k0, hi);
    const v16bf af1 = ld_a_frag(a1p + k0, hi);
#pragma unroll
    for (int nt = 0; nt < 4; ++nt) {
      const __bf16* bp = Bt + (long long)(col0 + nt * 16 + r) * ldb + k0;
      const v16bf bfrag = ld_a_frag(bp, hi);
      acc[0][nt] = __builtin_amdgcn_wmma_f32_16x16x32_bf16(
          false, af0, false, bfrag, (short)0, acc[0][nt], false, false);
      acc[1][nt] = __builtin_amdgcn_wmma_f32_16x16x32_bf16(
          false, af1, false, bfrag, (short)0, acc[1][nt], false, false);
    }
  }

  // epilogue: acc[fr][nt] elem i -> row row0+fr*16+hi*8+i, col col0+nt*16+r
#pragma unroll
  for (int fr = 0; fr < 2; ++fr) {
#pragma unroll
    for (int nt = 0; nt < 4; ++nt) {
      const int col = col0 + nt * 16 + r;
      if (col >= N) continue;
      const float bv = bias ? bias[col] : 0.0f;
#pragma unroll
      for (int i = 0; i < 8; ++i) {
        const int row = row0 + fr * 16 + hi * 8 + i;
        float v = acc[fr][nt][i] + bv;
        if (act == 1) v = tanhf(v);
        if (Cf) Cf[(long long)row * ldcf + col] = v;
        if (Cb) Cb[(long long)row * ldcb + col] = f2bf(v);
      }
    }
  }
}

// ---------------------------------------------------------------------------
// One-time conversion / transpose kernels
// ---------------------------------------------------------------------------
__global__ void cvt_kernel(const float* __restrict__ src,
                           __bf16* __restrict__ dst, long long n)
{
  const long long i = ((long long)blockIdx.x * 256 + threadIdx.x) * 4;
  if (i >= n) return;
  const float4 v = *reinterpret_cast<const float4*>(src + i);
  dst[i]     = f2bf(v.x);
  dst[i + 1] = f2bf(v.y);
  dst[i + 2] = f2bf(v.z);
  dst[i + 3] = f2bf(v.w);
}

// dst[n*K + k] = bf16( src[k*N + n] )   (K x N  ->  N x K), batched over z
__global__ __launch_bounds__(256) void transpose_cvt_kernel(
    const float* __restrict__ src, __bf16* __restrict__ dst,
    int K, int N, long long sSrc, long long sDst)
{
  src += blockIdx.z * sSrc;
  dst += blockIdx.z * sDst;
  __shared__ float tile[32][33];
  const int k0 = blockIdx.y * 32, n0 = blockIdx.x * 32;
  const int tx = threadIdx.x & 31, ty = threadIdx.x >> 5;  // 32 x 8
  for (int i = ty; i < 32; i += 8)
    tile[i][tx] = src[(long long)(k0 + i) * N + (n0 + tx)];
  __syncthreads();
  for (int i = ty; i < 32; i += 8)
    dst[(long long)(n0 + i) * K + (k0 + tx)] = f2bf(tile[tx][i]);
}

// ---------------------------------------------------------------------------
// Scan kernels
// ---------------------------------------------------------------------------
__global__ void concat1_kernel(const float* __restrict__ trg,
                               const __bf16* __restrict__ att_all_bf,
                               __bf16* __restrict__ rnn_in, int t)
{
  const int i = blockIdx.x * 256 + threadIdx.x;
  const int W = EE + HH;
  if (i >= BB * W) return;
  const int b = i / W, c = i % W;
  __bf16 v;
  if (c < EE)      v = f2bf(trg[((long long)b * TT + t) * EE + c]);
  else if (t == 0) v = f2bf(0.0f);
  else             v = att_all_bf[((long long)b * TT + (t - 1)) * HH + (c - EE)];
  rnn_in[i] = v;
}

__global__ void gru_gate_kernel(const float* __restrict__ gi,
                                const float* __restrict__ gh,
                                float* __restrict__ h,
                                __bf16* __restrict__ h_bf)
{
  const int i = blockIdx.x * 256 + threadIdx.x;
  if (i >= BB * HH) return;
  const int b = i >> 10, j = i & (HH - 1);
  const float* gib = gi + (long long)b * 3 * HH;
  const float* ghb = gh + (long long)b * 3 * HH;
  const float ir = gib[j], iz = gib[HH + j], in_ = gib[2 * HH + j];
  const float hr = ghb[j], hz = ghb[HH + j], hn  = ghb[2 * HH + j];
  const float rr = 1.0f / (1.0f + __expf(-(ir + hr)));
  const float zz = 1.0f / (1.0f + __expf(-(iz + hz)));
  const float nn = tanhf(in_ + rr * hn);
  const float hv = (1.0f - zz) * nn + zz * h[i];
  h[i]    = hv;
  h_bf[i] = f2bf(hv);
}

// per-b block (1024 thr = 32 waves): e -> softmax -> ctx, writes hc=[h,ctx] bf16
__global__ __launch_bounds__(1024) void attention_kernel(
    const float* __restrict__ proj_keys, const float* __restrict__ qw,
    const float* __restrict__ v_att, const unsigned char* __restrict__ src_mask,
    const float* __restrict__ enc, const float* __restrict__ h,
    __bf16* __restrict__ hc_bf)
{
  const int b = blockIdx.x;
  const int tid = threadIdx.x;
  const int wave = tid >> 5, lane = tid & 31;
  __shared__ float e_s[SS];

  const float* q = qw + (long long)b * HH;
  for (int s = wave; s < SS; s += 32) {
    const float* pk = proj_keys + ((long long)b * SS + s) * HH;
    float p = 0.0f;
    for (int k = lane; k < HH; k += 32) p += tanhf(pk[k] + q[k]) * v_att[k];
    for (int off = 16; off; off >>= 1) p += __shfl_xor(p, off, 32);
    if (lane == 0) e_s[s] = p;
  }
  __syncthreads();
  if (tid == 0) {
    float mx = -1e30f;
    for (int s = 0; s < SS; ++s) {
      float v = src_mask[(long long)b * SS + s] ? e_s[s] : -1e9f;
      e_s[s] = v;  mx = fmaxf(mx, v);
    }
    float sum = 0.0f;
    for (int s = 0; s < SS; ++s) { float ex = __expf(e_s[s] - mx); e_s[s] = ex; sum += ex; }
    const float inv = 1.0f / sum;
    for (int s = 0; s < SS; ++s) e_s[s] *= inv;
  }
  __syncthreads();
  float c = 0.0f;
  const float* eb = enc + (long long)b * SS * HH;
  for (int s = 0; s < SS; ++s) c += e_s[s] * eb[(long long)s * HH + tid];
  hc_bf[(long long)b * 2 * HH + tid]      = f2bf(h[(long long)b * HH + tid]);
  hc_bf[(long long)b * 2 * HH + HH + tid] = f2bf(c);
}

// per-b block: 4 KB-attention score rows (idx 0..3), hop chaining, cache update
__global__ __launch_bounds__(1024) void kbscore_kernel(
    const float* __restrict__ proj_kb,   // (4,B,KD,H)
    const float* __restrict__ qout,      // (5,B,H); idx -> slice 1+idx
    const float* __restrict__ v_kb,      // (4,H)
    float* __restrict__ cache,           // (2,B,KD)
    float* __restrict__ u_small,         // (B,T,2,KD)
    int t)
{
  const int b = blockIdx.x;
  const int tid = threadIdx.x;
  const int wave = tid >> 5, lane = tid & 31;
  __shared__ float raw_s[NATT * KDc];

  for (int rt = wave; rt < NATT * KDc; rt += 32) {
    const int idx = rt >> 6, k = rt & (KDc - 1);
    const float* pk = proj_kb + (((long long)idx * BB + b) * KDc + k) * HH;
    const float* q  = qout + ((long long)(1 + idx) * BB + b) * HH;
    const float* v  = v_kb + (long long)idx * HH;
    float p = 0.0f;
    for (int hh = lane; hh < HH; hh += 32) p += tanhf(pk[hh] + q[hh]) * v[hh];
    for (int off = 16; off; off >>= 1) p += __shfl_xor(p, off, 32);
    if (lane == 0) raw_s[rt] = p;
  }
  __syncthreads();
  if (tid < 2 * KDc) {
    const int m = tid >> 6, k = tid & (KDc - 1);
    const long long ci = ((long long)m * BB + b) * KDc + k;
    const float mid = raw_s[m * KDc + k] + cache[ci];          // hop 0
    const float out = raw_s[(2 + m) * KDc + k] + mid;          // hop 1
    cache[ci] = out;
    u_small[(((long long)b * TT + t) * 2 + m) * KDc + k] = out;
  }
}

// out[b,t,kb_values[b,p]] += (~km) * (r0[b,t,p/64] + r1[b,t,p%64])
__global__ void scatter_kernel(const float* __restrict__ u_small,
                               const unsigned char* __restrict__ kb_mask,
                               const int* __restrict__ kb_values,
                               float* __restrict__ out)
{
  const long long i = (long long)blockIdx.x * 256 + threadIdx.x;
  if (i >= (long long)BB * TT * KBT) return;
  const int p = (int)(i & (KBT - 1));
  const long long bt = i >> 12;                  // KBT = 2^12
  const int b = (int)(bt >> 6);                  // TT  = 2^6
  if (kb_mask[(long long)b * KBT + p]) return;
  const float val = u_small[(bt * 2 + 0) * KDc + (p >> 6)]
                  + u_small[(bt * 2 + 1) * KDc + (p & 63)];
  const int v = kb_values[(long long)b * KBT + p];
  atomicAdd(out + bt * VV + v, val);
}

// ---------------------------------------------------------------------------
// Host side
// ---------------------------------------------------------------------------
static void gemm(hipStream_t s, const __bf16* A, const __bf16* Bt,
                 const float* bias, float* Cf, __bf16* Cb,
                 int M, int N, int K, int lda, int ldb, int ldcf, int ldcb,
                 int act, long long sA = 0, long long sB = 0, long long sC = 0,
                 int batch = 1)
{
  dim3 grid(N / 128, (M + 63) / 64, batch);
  wmma_gemm_bf16_kernel<<<grid, 128, 0, s>>>(A, Bt, bias, Cf, Cb, M, N, K,
                                             lda, ldb, ldcf, ldcb, act,
                                             sA, sB, sC);
}

static void cvt(hipStream_t s, const float* src, __bf16* dst, long long n) {
  cvt_kernel<<<(unsigned)((n / 4 + 255) / 256), 256, 0, s>>>(src, dst, n);
}

static void tcvt(hipStream_t s, const float* src, __bf16* dst, int K, int N,
                 long long sSrc = 0, long long sDst = 0, int batch = 1) {
  dim3 grid(N / 32, K / 32, batch);
  transpose_cvt_kernel<<<grid, 256, 0, s>>>(src, dst, K, N, sSrc, sDst);
}

extern "C" void kernel_launch(void* const* d_in, const int* in_sizes, int n_in,
                              void* d_out, int out_size, void* d_ws, size_t ws_size,
                              hipStream_t stream)
{
  (void)in_sizes; (void)n_in; (void)out_size; (void)ws_size;

  const float* trg_embed  = (const float*)d_in[0];   // (B,T,E)
  const float* enc_out    = (const float*)d_in[1];   // (B,S,H)
  const float* enc_hidden = (const float*)d_in[2];   // (B,H)
  const float* kb_keys    = (const float*)d_in[3];   // (2,B,KD,DKB)
  const unsigned char* src_mask = (const unsigned char*)d_in[4]; // (B,1,S)
  const unsigned char* kb_mask  = (const unsigned char*)d_in[5]; // (B,1,KBT)
  const int*   kb_values  = (const int*)d_in[6];     // (B,KBT)
  const float* W_ih  = (const float*)d_in[7];        // (3H, E+H)
  const float* W_hh  = (const float*)d_in[8];        // (3H, H)
  const float* b_ih  = (const float*)d_in[9];
  const float* b_hh  = (const float*)d_in[10];
  const float* W_bridge = (const float*)d_in[11];    // (H,H)
  const float* b_bridge = (const float*)d_in[12];
  const float* W_key    = (const float*)d_in[13];    // (H,H)
  const float* W_query  = (const float*)d_in[14];    // (H,H)
  const float* v_att    = (const float*)d_in[15];    // (H)
  const float* W_av     = (const float*)d_in[16];    // (2H,H)
  const float* b_av     = (const float*)d_in[17];
  const float* W_out    = (const float*)d_in[18];    // (H,V)
  const float* b_out    = (const float*)d_in[19];
  const float* Wk_kb    = (const float*)d_in[20];    // (4,DKB,H)
  const float* Wq_kb    = (const float*)d_in[21];    // (4,H,H)
  const float* v_kb     = (const float*)d_in[22];    // (4,H)

  float* out = (float*)d_out;                        // (B,T,V)

  // ---- workspace carve (256B aligned)
  char* w = (char*)d_ws;
  auto carveF = [&](size_t elems) {
    float* p = (float*)w;  w += (elems * 4 + 255) & ~(size_t)255;  return p;
  };
  auto carveB = [&](size_t elems) {
    __bf16* p = (__bf16*)w;  w += (elems * 2 + 255) & ~(size_t)255;  return p;
  };
  // f32 scratch
  float* proj_keys = carveF((size_t)BB * SS * HH);           // 32 MB
  float* proj_kb   = carveF((size_t)NATT * BB * KDc * HH);   // 64 MB
  float* gi        = carveF((size_t)BB * 3 * HH);
  float* gh        = carveF((size_t)BB * 3 * HH);
  float* qout      = carveF((size_t)5 * BB * HH);
  float* hbuf      = carveF((size_t)BB * HH);
  float* cache     = carveF((size_t)2 * BB * KDc);
  float* u_small   = carveF((size_t)BB * TT * 2 * KDc);      // 2 MB
  // bf16 operands
  __bf16* enc_hid_bf = carveB((size_t)BB * HH);
  __bf16* enc_out_bf = carveB((size_t)BB * SS * HH);         // 16.8 MB
  __bf16* kb_keys_bf = carveB((size_t)2 * BB * KDc * DKB);   // 8.4 MB
  __bf16* W_ih_bf    = carveB((size_t)3 * HH * (EE + HH));   // 9.4 MB
  __bf16* W_hh_bf    = carveB((size_t)3 * HH * HH);          // 6.3 MB
  __bf16* Wbr_t      = carveB((size_t)HH * HH);
  __bf16* Wkey_t     = carveB((size_t)HH * HH);
  __bf16* Wq5_t      = carveB((size_t)5 * HH * HH);          // 10.5 MB
  __bf16* Wav_t      = carveB((size_t)HH * 2 * HH);          // 4.2 MB
  __bf16* Wkkb_t     = carveB((size_t)NATT * HH * DKB);      // 4.2 MB
  __bf16* Wout_t     = carveB((size_t)VV * HH);              // 65.5 MB
  __bf16* rnn_in_bf  = carveB((size_t)BB * (EE + HH));
  __bf16* h_bf       = carveB((size_t)BB * HH);
  __bf16* hc_bf      = carveB((size_t)BB * 2 * HH);
  __bf16* att_all_bf = carveB((size_t)BB * TT * HH);         // 8.4 MB

  // ---- prelude: one-time conversions / transposes ------------------------
  hipMemsetAsync(cache, 0, (size_t)2 * BB * KDc * 4, stream);

  cvt(stream, enc_hidden, enc_hid_bf, (long long)BB * HH);
  cvt(stream, enc_out,    enc_out_bf, (long long)BB * SS * HH);
  cvt(stream, kb_keys,    kb_keys_bf, (long long)2 * BB * KDc * DKB);
  cvt(stream, W_ih,       W_ih_bf,    (long long)3 * HH * (EE + HH)); // already (N,K)
  cvt(stream, W_hh,       W_hh_bf,    (long long)3 * HH * HH);        // already (N,K)
  tcvt(stream, W_bridge, Wbr_t,  HH, HH);
  tcvt(stream, W_key,    Wkey_t, HH, HH);
  tcvt(stream, W_query,  Wq5_t,  HH, HH);                              // slot 0
  tcvt(stream, Wq_kb, Wq5_t + (size_t)HH * HH, HH, HH,
       (long long)HH * HH, (long long)HH * HH, NATT);                  // slots 1..4
  tcvt(stream, W_av,  Wav_t, 2 * HH, HH);
  tcvt(stream, Wk_kb, Wkkb_t, DKB, HH,
       (long long)DKB * HH, (long long)HH * DKB, NATT);
  tcvt(stream, W_out, Wout_t, HH, VV);                                 // (V,H)

  // h0 = tanh(enc_hidden @ W_bridge + b_bridge)  -> hbuf (f32) + h_bf
  gemm(stream, enc_hid_bf, Wbr_t, b_bridge, hbuf, h_bf,
       BB, HH, HH, HH, HH, HH, HH, 1);
  // proj_keys = enc_out @ W_key
  gemm(stream, enc_out_bf, Wkey_t, nullptr, proj_keys, nullptr,
       BB * SS, HH, HH, HH, HH, HH, 0, 0);
  // proj_kb[i] = kb_keys[i%2] @ Wk_kb[i]
  for (int i = 0; i < NATT; ++i)
    gemm(stream, kb_keys_bf + (size_t)(i & 1) * BB * KDc * DKB,
         Wkkb_t + (size_t)i * HH * DKB, nullptr,
         proj_kb + (size_t)i * BB * KDc * HH, nullptr,
         BB * KDc, HH, DKB, DKB, DKB, HH, 0, 0);

  // ---- sequential scan ---------------------------------------------------
  for (int t = 0; t < TT; ++t) {
    {
      int n = BB * (EE + HH);
      concat1_kernel<<<(n + 255) / 256, 256, 0, stream>>>(trg_embed, att_all_bf,
                                                          rnn_in_bf, t);
    }
    // gi = rnn_in @ W_ih^T + b_ih   (W_ih already (N,K))
    gemm(stream, rnn_in_bf, W_ih_bf, b_ih, gi, nullptr,
         BB, 3 * HH, EE + HH, EE + HH, EE + HH, 3 * HH, 0, 0);
    // gh = h @ W_hh^T + b_hh
    gemm(stream, h_bf, W_hh_bf, b_hh, gh, nullptr,
         BB, 3 * HH, HH, HH, HH, 3 * HH, 0, 0);
    {
      int n = BB * HH;
      gru_gate_kernel<<<(n + 255) / 256, 256, 0, stream>>>(gi, gh, hbuf, h_bf);
    }
    // qout[z] = h @ Wq5[z]   (z=0: W_query, 1..4: Wq_kb)
    gemm(stream, h_bf, Wq5_t, nullptr, qout, nullptr,
         BB, HH, HH, HH, HH, HH, 0, 0,
         0, (long long)HH * HH, (long long)BB * HH, 5);
    attention_kernel<<<BB, 1024, 0, stream>>>(proj_keys, qout, v_att, src_mask,
                                              enc_out, hbuf, hc_bf);
    kbscore_kernel<<<BB, 1024, 0, stream>>>(proj_kb, qout, v_kb, cache, u_small, t);
    // att_v_t = tanh([h,ctx] @ W_av + b_av) -> bf16 into att_all_bf[:, t, :]
    gemm(stream, hc_bf, Wav_t, b_av, nullptr, att_all_bf + (size_t)t * HH,
         BB, HH, 2 * HH, 2 * HH, 2 * HH, 0, TT * HH, 1);
  }

  // ---- output projection (dominant GEMM, 268 GFLOP) + scatter-add --------
  gemm(stream, att_all_bf, Wout_t, b_out, out, nullptr,
       BB * TT, VV, HH, HH, HH, VV, 0, 0);
  {
    long long n = (long long)BB * TT * KBT;
    scatter_kernel<<<(unsigned)((n + 255) / 256), 256, 0, stream>>>(
        u_small, kb_mask, kb_values, out);
  }
}
